// TriangleMultiplicationOutgoing_12103217840565
// MI455X (gfx1250) — matrix operand order repaired and checked by reading, not compiled, not run on hardware
//
#include <hip/hip_runtime.h>
#include <hip/hip_bf16.h>

// ---------------------------------------------------------------------------
// TriangleMultiplicationOutgoing: N=512, C=128, all-f16 WMMA pipeline
// ---------------------------------------------------------------------------
#define CZ   128
#define NSP  512
#define MTOT (NSP * NSP)   // 262144 flattened pixels
#define WSZ  (CZ * CZ)     // 16384 elements per weight matrix

typedef __attribute__((ext_vector_type(16))) _Float16 v16h;
typedef __attribute__((ext_vector_type(8)))  float    v8f;

union U16x8  { uint4 u;    _Float16 h[8]; };
union U16x16 { uint4 u[2]; v16h h; };

__device__ __forceinline__ v8f wmma_f16(v16h a, v16h b, v8f c) {
  // D = A(16x32,f16) * B(32x16,f16) + C(16x16,f32)
  return __builtin_amdgcn_wmma_f32_16x16x32_f16(
      /*neg_a=*/false, a, /*neg_b=*/false, b,
      /*c_mod=*/(short)0, c, /*reuse_a=*/false, /*reuse_b=*/false);
}

__device__ __forceinline__ float sigmoidf_fast(float x) {
  return 1.0f / (1.0f + __expf(-x));
}

// CDNA5 async copy: 16B global -> LDS, tracked by ASYNCcnt.
__device__ __forceinline__ void async_b128_to_lds(_Float16* lds,
                                                  const _Float16* gptr) {
  asm volatile("global_load_async_to_lds_b128 %0, %1, off"
               :: "v"((unsigned)(unsigned long long)lds),
                  "v"((unsigned long long)gptr)
               : "memory");
}
__device__ __forceinline__ void wait_async0() {
  asm volatile("s_wait_asynccnt 0x0" ::: "memory");
}

// A fragment: 16x32 f16 row-major (leading dim ld, 16B-aligned rows).
// lane m=lane&15; lanes<16 hold K 0..15, lanes>=16 K 16..31 (contiguous, matches B)
// -> one 32B run => two (ds_)load_b128.
__device__ __forceinline__ v16h load_frag_a(const _Float16* __restrict__ As,
                                            int ld, int lane) {
  const int m  = lane & 15;
  const int kb = ((lane >> 4) & 1) << 4;   // 0 or 16
  U16x16 u;
  u.u[0] = *(const uint4*)(As + m * ld + kb);
  u.u[1] = *(const uint4*)(As + m * ld + kb + 8);
  return u.h;
}

// B fragment for C = A*B^T, B^T row-major [n][k]: lane n=lane&15, 16 contiguous K
// (lanes<16: K 0..15, lanes>=16: K 16..31) => one 32B run, two b128 loads.
__device__ __forceinline__ v16h load_frag_b(const _Float16* __restrict__ Bs,
                                            int ld, int lane) {
  const int n  = lane & 15;
  const int kb = ((lane >> 4) & 1) << 4;   // 0 or 16
  U16x16 u;
  u.u[0] = *(const uint4*)(Bs + n * ld + kb);
  u.u[1] = *(const uint4*)(Bs + n * ld + kb + 8);
  return u.h;
}

// B fragment from a pre-converted f16 weight W[out][in] (row n = output channel).
__device__ __forceinline__ v16h frag_b_weight16(const _Float16* __restrict__ Wh,
                                                int nt, int ks, int lane) {
  const int n  = nt * 16 + (lane & 15);
  const int kb = ks * 32 + (((lane >> 4) & 1) << 4);
  U16x16 u;
  u.u[0] = *(const uint4*)(Wh + n * CZ + kb);
  u.u[1] = *(const uint4*)(Wh + n * CZ + kb + 8);
  return u.h;
}

__device__ __forceinline__ v8f frag_bias(const float* __restrict__ b,
                                         int nt, int lane) {
  const float bv = b[nt * 16 + (lane & 15)];
  v8f c;
#pragma unroll
  for (int v = 0; v < 8; ++v) c[v] = bv;
  return c;
}

// ---------------------------------------------------------------------------
// Kernel 0: one-shot fp32 -> f16 conversion of the six weight matrices.
// wh layout: [0]=Wa [1]=Wga [2]=Wb [3]=Wgb [4]=Wog [5]=Wo
// ---------------------------------------------------------------------------
__global__ __launch_bounds__(256) void k0_cvt_weights(
    const float* __restrict__ W0, const float* __restrict__ W1,
    const float* __restrict__ W2, const float* __restrict__ W3,
    const float* __restrict__ W4, const float* __restrict__ W5,
    _Float16* __restrict__ wh)
{
  const float* Ws[6] = {W0, W1, W2, W3, W4, W5};
  const int wsel = blockIdx.y;
  const int i = blockIdx.x * 256 + threadIdx.x;   // 64 blocks * 256 = 16384
  wh[wsel * WSZ + i] = (_Float16)Ws[wsel][i];
}

// ---------------------------------------------------------------------------
// Kernel 1: fused layernorm + 2 gated projections + out-gate.
// 256 threads (8 waves), 64 pixel-rows. Wave w owns output-channel tile nt=w;
// each weight fragment is reused across all 4 row tiles. a,b staged in LDS
// channel-major and flushed 16B-coalesced to aT[c][pixel], bT[c][pixel].
// ---------------------------------------------------------------------------
#define ZN_LD 136   // padded f16 ld: 272B row stride -> conflict-free b128 gathers
#define ST_LD 72    // staging ld: 144B stride

__global__ __launch_bounds__(256) void k1_ln_gated_proj(
    const float* __restrict__ z, const float* __restrict__ gamma,
    const float* __restrict__ beta,
    const float* __restrict__ ba,  const float* __restrict__ bga,
    const float* __restrict__ bb,  const float* __restrict__ bgb,
    const float* __restrict__ bog,
    const _Float16* __restrict__ wh,
    _Float16* __restrict__ aT, _Float16* __restrict__ bT,
    _Float16* __restrict__ og)
{
  __shared__ float zf[64 * CZ];                         // 32 KB
  __shared__ __align__(16) _Float16 zn[64 * ZN_LD];     // 17 KB
  __shared__ float red[64][8];
  __shared__ __align__(16) _Float16 aStage[CZ * ST_LD]; // 18 KB
  __shared__ __align__(16) _Float16 bStage[CZ * ST_LD]; // 18 KB

  const int  t       = threadIdx.x;
  const long rowbase = (long)blockIdx.x * 64;

  const _Float16* Wa  = wh + 0 * WSZ;
  const _Float16* Wga = wh + 1 * WSZ;
  const _Float16* Wb  = wh + 2 * WSZ;
  const _Float16* Wgb = wh + 3 * WSZ;
  const _Float16* Wog = wh + 4 * WSZ;

  if (t < CZ) {   // warm WGP$/L2 with the f16 weights (global_prefetch_b8)
    __builtin_prefetch(Wa  + t * CZ, 0, 1);
    __builtin_prefetch(Wgb + t * CZ, 0, 1);
  }

  // ---- stage 1: coalesced 64x128 fp32 tile load
  const float* zsrc = z + rowbase * (long)CZ;
#pragma unroll
  for (int i = 0; i < 32; ++i) zf[t + 256 * i] = zsrc[t + 256 * i];
  __syncthreads();

  // ---- stage 2: layernorm, 4 threads per row
  const int r = t >> 2, q = t & 3;
  float s = 0.f, ss = 0.f;
#pragma unroll
  for (int k = 0; k < 32; ++k) {
    float v = zf[r * CZ + q * 32 + k];
    s += v; ss += v * v;
  }
  red[r][q] = s; red[r][4 + q] = ss;
  __syncthreads();
  const float mean = (red[r][0] + red[r][1] + red[r][2] + red[r][3]) * (1.0f / CZ);
  const float var  = (red[r][4] + red[r][5] + red[r][6] + red[r][7]) * (1.0f / CZ)
                     - mean * mean;
  const float rstd = rsqrtf(var + 1e-5f);
#pragma unroll
  for (int k = 0; k < 32; ++k) {
    const int c = q * 32 + k;
    const float v = (zf[r * CZ + c] - mean) * rstd * gamma[c] + beta[c];
    zn[r * ZN_LD + c] = (_Float16)v;
  }
  __syncthreads();

  // ---- stage 3: WMMA projections; wave w -> channel tile nt=w, mt=0..3
  const int w = t >> 5, lane = t & 31;
  const int n  = lane & 15;
  const int mh = ((lane >> 4) & 1) << 3;
  const int nt = w;
  const int c  = nt * 16 + n;

  // gated projection A -> aStage[c][row]
  {
    v8f accg[4], accp[4];
#pragma unroll
    for (int mt = 0; mt < 4; ++mt) {
      accg[mt] = frag_bias(bga, nt, lane);
      accp[mt] = frag_bias(ba,  nt, lane);
    }
#pragma unroll
    for (int ks = 0; ks < 4; ++ks) {
      const v16h bg = frag_b_weight16(Wga, nt, ks, lane);
      const v16h bp = frag_b_weight16(Wa,  nt, ks, lane);
#pragma unroll
      for (int mt = 0; mt < 4; ++mt) {
        const v16h af = load_frag_a(zn + mt * 16 * ZN_LD + ks * 32, ZN_LD, lane);
        accg[mt] = wmma_f16(af, bg, accg[mt]);
        accp[mt] = wmma_f16(af, bp, accp[mt]);
      }
    }
#pragma unroll
    for (int mt = 0; mt < 4; ++mt)
#pragma unroll
      for (int v = 0; v < 8; ++v)
        aStage[c * ST_LD + mt * 16 + mh + v] =
            (_Float16)(sigmoidf_fast(accg[mt][v]) * accp[mt][v]);
  }
  // gated projection B -> bStage[c][row]
  {
    v8f accg[4], accp[4];
#pragma unroll
    for (int mt = 0; mt < 4; ++mt) {
      accg[mt] = frag_bias(bgb, nt, lane);
      accp[mt] = frag_bias(bb,  nt, lane);
    }
#pragma unroll
    for (int ks = 0; ks < 4; ++ks) {
      const v16h bg = frag_b_weight16(Wgb, nt, ks, lane);
      const v16h bp = frag_b_weight16(Wb,  nt, ks, lane);
#pragma unroll
      for (int mt = 0; mt < 4; ++mt) {
        const v16h af = load_frag_a(zn + mt * 16 * ZN_LD + ks * 32, ZN_LD, lane);
        accg[mt] = wmma_f16(af, bg, accg[mt]);
        accp[mt] = wmma_f16(af, bp, accp[mt]);
      }
    }
#pragma unroll
    for (int mt = 0; mt < 4; ++mt)
#pragma unroll
      for (int v = 0; v < 8; ++v)
        bStage[c * ST_LD + mt * 16 + mh + v] =
            (_Float16)(sigmoidf_fast(accg[mt][v]) * accp[mt][v]);
  }
  // out gate -> og[pixel][c] (f16, lane-coalesced)
  {
    v8f acc[4];
#pragma unroll
    for (int mt = 0; mt < 4; ++mt) acc[mt] = frag_bias(bog, nt, lane);
#pragma unroll
    for (int ks = 0; ks < 4; ++ks) {
      const v16h bf = frag_b_weight16(Wog, nt, ks, lane);
#pragma unroll
      for (int mt = 0; mt < 4; ++mt) {
        const v16h af = load_frag_a(zn + mt * 16 * ZN_LD + ks * 32, ZN_LD, lane);
        acc[mt] = wmma_f16(af, bf, acc[mt]);
      }
    }
#pragma unroll
    for (int mt = 0; mt < 4; ++mt)
#pragma unroll
      for (int v = 0; v < 8; ++v) {
        const long row = rowbase + mt * 16 + mh + v;
        og[row * CZ + c] = (_Float16)sigmoidf_fast(acc[mt][v]);
      }
  }
  __syncthreads();

  // ---- stage 4: coalesced channel-major flush (16B stores)
  const int cc = t >> 1;
  const int r0 = (t & 1) * 32;
#pragma unroll
  for (int i = 0; i < 4; ++i) {
    const int off = r0 + i * 8;
    *(uint4*)(aT + (long)cc * MTOT + rowbase + off) =
        *(const uint4*)(aStage + cc * ST_LD + off);
    *(uint4*)(bT + (long)cc * MTOT + rowbase + off) =
        *(const uint4*)(bStage + cc * ST_LD + off);
  }
}

// ---------------------------------------------------------------------------
// Kernel 2: per-channel triangle GEMM  U_c = A_c * B_c^T  (512x512x512, f16).
// 128x128 output tile per block; wave w owns row tile mt=w and all 8 nt, so
// one A fragment feeds 8 WMMAs per K-slab. Double-buffered async global->LDS
// K-slabs (ASYNCcnt). Result staged in LDS, flushed channel-major coalesced.
// ---------------------------------------------------------------------------
#define KT_LD  40
#define OST_LD 136

__global__ __launch_bounds__(256) void k2_triangle(
    const _Float16* __restrict__ aT, const _Float16* __restrict__ bT,
    _Float16* __restrict__ upd)
{
  __shared__ __align__(16) _Float16 As[2][128 * KT_LD];  // 2 x 10 KB
  __shared__ __align__(16) _Float16 Bs[2][128 * KT_LD];  // 2 x 10 KB
  __shared__ __align__(16) _Float16 Ost[128 * OST_LD];   // 34 KB

  const int c  = blockIdx.y;                 // channel 0..127
  const int it = blockIdx.x >> 2;            // 4x4 grid of 128x128 tiles
  const int jt = blockIdx.x & 3;
  const _Float16* A = aT + (long)c * MTOT + (long)(it * 128) * NSP;
  const _Float16* B = bT + (long)c * MTOT + (long)(jt * 128) * NSP;

  const int t = threadIdx.x, w = t >> 5, lane = t & 31;
  const int mt = w;                          // wave's row tile (16 rows)

  v8f acc[8];
#pragma unroll
  for (int nt = 0; nt < 8; ++nt)
#pragma unroll
    for (int v = 0; v < 8; ++v) acc[nt][v] = 0.f;

  // 128x32 f16 slab = 512 chunks of 16B; 2 A-chunks + 2 B-chunks per thread
  const int cid0 = t, cid1 = t + 256;
  const int lr0 = cid0 >> 2, lc0 = (cid0 & 3) * 8;
  const int lr1 = cid1 >> 2, lc1 = (cid1 & 3) * 8;

  // prime buffer 0
  async_b128_to_lds(&As[0][lr0 * KT_LD + lc0], A + (long)lr0 * NSP + lc0);
  async_b128_to_lds(&As[0][lr1 * KT_LD + lc1], A + (long)lr1 * NSP + lc1);
  async_b128_to_lds(&Bs[0][lr0 * KT_LD + lc0], B + (long)lr0 * NSP + lc0);
  async_b128_to_lds(&Bs[0][lr1 * KT_LD + lc1], B + (long)lr1 * NSP + lc1);

  int cur = 0;
  for (int kk = 0; kk < NSP; kk += 32) {
    wait_async0();          // our slab writes done (per-wave), then
    __syncthreads();        // barrier publishes every wave's slab
    if (kk + 32 < NSP) {    // prefetch next slab into the other buffer
      const int kn = kk + 32;
      async_b128_to_lds(&As[cur ^ 1][lr0 * KT_LD + lc0], A + (long)lr0 * NSP + kn + lc0);
      async_b128_to_lds(&As[cur ^ 1][lr1 * KT_LD + lc1], A + (long)lr1 * NSP + kn + lc1);
      async_b128_to_lds(&Bs[cur ^ 1][lr0 * KT_LD + lc0], B + (long)lr0 * NSP + kn + lc0);
      async_b128_to_lds(&Bs[cur ^ 1][lr1 * KT_LD + lc1], B + (long)lr1 * NSP + kn + lc1);
    }
    const v16h af = load_frag_a(&As[cur][mt * 16 * KT_LD], KT_LD, lane);
#pragma unroll
    for (int nt = 0; nt < 8; ++nt) {
      const v16h bf = load_frag_b(&Bs[cur][nt * 16 * KT_LD], KT_LD, lane);
      acc[nt] = wmma_f16(af, bf, acc[nt]);   // A fragment reused 8x
    }
    cur ^= 1;
  }

  // stage 128x128 result tile, then flush channel-major coalesced
  const int n  = lane & 15;
  const int mh = ((lane >> 4) & 1) << 3;
#pragma unroll
  for (int nt = 0; nt < 8; ++nt)
#pragma unroll
    for (int v = 0; v < 8; ++v)
      Ost[(mt * 16 + mh + v) * OST_LD + nt * 16 + n] = (_Float16)acc[nt][v];
  __syncthreads();

  _Float16* U = upd + (long)c * MTOT;        // channel-major update[c][i*512+j]
#pragma unroll
  for (int i = 0; i < 8; ++i) {
    const int cid = t + 256 * i;             // 2048 chunks of 16B
    const int rr  = cid >> 4;
    const int ch  = (cid & 15) * 8;
    *(uint4*)(U + (long)(it * 128 + rr) * NSP + jt * 128 + ch) =
        *(const uint4*)(Ost + rr * OST_LD + ch);
  }
}

// ---------------------------------------------------------------------------
// Kernel 3: out = z + og * (update Wo^T + bo).
// Channel-major update transposed through LDS on load (coalesced reads).
// Wave w owns channel tile nt=w; Wo fragments reused across 4 row tiles.
// ---------------------------------------------------------------------------
__global__ __launch_bounds__(256) void k3_out_proj(
    const _Float16* __restrict__ upd, const _Float16* __restrict__ og,
    const float* __restrict__ z, const _Float16* __restrict__ wh,
    const float* __restrict__ bo, float* __restrict__ out)
{
  __shared__ __align__(16) _Float16 Us[64 * ZN_LD];

  const int  t       = threadIdx.x;
  const long rowbase = (long)blockIdx.x * 64;
  const _Float16* Wo = wh + 5 * WSZ;

  if (t < CZ) __builtin_prefetch(Wo + t * CZ, 0, 1);

  // load channel-major update: per channel, 64 contiguous pixels (coalesced),
  // transpose into Us[row][c]
  const int cc = t >> 1;
  const int r0 = (t & 1) * 32;
#pragma unroll
  for (int i = 0; i < 4; ++i) {
    U16x8 u;
    u.u = *(const uint4*)(upd + (long)cc * MTOT + rowbase + r0 + i * 8);
#pragma unroll
    for (int e = 0; e < 8; ++e)
      Us[(r0 + i * 8 + e) * ZN_LD + cc] = u.h[e];
  }
  __syncthreads();

  const int w = t >> 5, lane = t & 31;
  const int n  = lane & 15;
  const int mh = ((lane >> 4) & 1) << 3;
  const int nt = w;

  v8f acc[4];
#pragma unroll
  for (int mt = 0; mt < 4; ++mt) acc[mt] = frag_bias(bo, nt, lane);
#pragma unroll
  for (int ks = 0; ks < 4; ++ks) {
    const v16h bf = frag_b_weight16(Wo, nt, ks, lane);
#pragma unroll
    for (int mt = 0; mt < 4; ++mt) {
      const v16h af = load_frag_a(Us + mt * 16 * ZN_LD + ks * 32, ZN_LD, lane);
      acc[mt] = wmma_f16(af, bf, acc[mt]);
    }
  }
#pragma unroll
  for (int mt = 0; mt < 4; ++mt)
#pragma unroll
    for (int v = 0; v < 8; ++v) {
      const long row = rowbase + mt * 16 + mh + v;
      const long idx = row * CZ + nt * 16 + n;
      out[idx] = z[idx] + (float)og[idx] * acc[mt][v];
    }
}

// ---------------------------------------------------------------------------
extern "C" void kernel_launch(void* const* d_in, const int* in_sizes, int n_in,
                              void* d_out, int out_size, void* d_ws, size_t ws_size,
                              hipStream_t stream) {
  const float* z     = (const float*)d_in[0];
  const float* gamma = (const float*)d_in[1];
  const float* beta  = (const float*)d_in[2];
  const float* Wa  = (const float*)d_in[3];   const float* ba  = (const float*)d_in[4];
  const float* Wb  = (const float*)d_in[5];   const float* bb  = (const float*)d_in[6];
  const float* Wga = (const float*)d_in[7];   const float* bga = (const float*)d_in[8];
  const float* Wgb = (const float*)d_in[9];   const float* bgb = (const float*)d_in[10];
  const float* Wo  = (const float*)d_in[11];  const float* bo  = (const float*)d_in[12];
  const float* Wog = (const float*)d_in[13];  const float* bog = (const float*)d_in[14];
  float* out = (float*)d_out;

  // workspace (f16): aT | bT | og | upd (4 x 64MB) | wh (192KB)
  char* ws = (char*)d_ws;
  const size_t seg = (size_t)CZ * MTOT * sizeof(_Float16);
  _Float16* aT  = (_Float16*)(ws + 0 * seg);
  _Float16* bT  = (_Float16*)(ws + 1 * seg);
  _Float16* og  = (_Float16*)(ws + 2 * seg);
  _Float16* upd = (_Float16*)(ws + 3 * seg);
  _Float16* wh  = (_Float16*)(ws + 4 * seg);

  k0_cvt_weights<<<dim3(64, 6), 256, 0, stream>>>(Wa, Wga, Wb, Wgb, Wog, Wo, wh);

  k1_ln_gated_proj<<<MTOT / 64, 256, 0, stream>>>(
      z, gamma, beta, ba, bga, bb, bgb, bog, wh, aT, bT, og);

  k2_triangle<<<dim3(16, CZ), 256, 0, stream>>>(aT, bT, upd);

  k3_out_proj<<<MTOT / 64, 256, 0, stream>>>(upd, og, z, wh, bo, out);
}
